// EncodingBlock_Job_21414706937964
// MI455X (gfx1250) — compile-verified, matrix operand order checked by reading, not run on hardware
//
#include <hip/hip_runtime.h>
#include <hip/hip_bf16.h>
#include <cstdint>

// ---------- types for WMMA ----------
typedef __attribute__((ext_vector_type(16))) __bf16 v16bf;
typedef __attribute__((ext_vector_type(8)))  float  v8f;
typedef __attribute__((ext_vector_type(4)))  unsigned int u32x4;

union BfFrag { v16bf v; u32x4 q[2]; };

// ---------- problem constants ----------
#define NB   32
#define NR   200
#define NMM  60
#define NVV  40
#define NC   100
#define NE   256
#define NH   16
#define ND   16
#define NHD  256
#define NF   512
#define RP   208      // R padded to 16
#define CP   112      // C padded to 16
#define CW   128      // C padded to 32 (WMMA K)
#define BH   (NB*NH)  // 512
#define BR   (NB*NR)  // 6400 rows
#define BC   (NB*NC)  // 3200 rows

// ---------- CDNA5 async global->LDS (ASYNCcnt path), inline asm ----------
__device__ __forceinline__ void async_load_b128(unsigned lds_off, const __bf16* gptr) {
  unsigned long long ga = (unsigned long long)(uintptr_t)gptr;
  asm volatile("global_load_async_to_lds_b128 %0, %1, off"
               :: "v"(lds_off), "v"(ga) : "memory");
}
__device__ __forceinline__ void wait_async_le3() {
  asm volatile("s_wait_asynccnt 0x3" ::: "memory");
}
__device__ __forceinline__ void wait_async_le0() {
  asm volatile("s_wait_asynccnt 0x0" ::: "memory");
}

// =====================================================================
// V1: direct-load batched WMMA GEMM (used for the small attention GEMMs)
// out[M,N] = alpha * A[M,K] @ (BT[N,K])^T + bias ; one wave = one 16x16 tile
// =====================================================================
__global__ __launch_bounds__(128) void gemm_wmma_bf16(
    const __bf16* __restrict__ A,  int lda, long long sA,
    const __bf16* __restrict__ BT, int ldb, long long sB,
    float*  __restrict__ outf, int ldo,  long long sO,
    __bf16* __restrict__ outb, int ldob, long long sOb,
    const float* __restrict__ bias,
    int Mtiles, int Ntiles, int kTiles,
    int Mreal, int Nreal, float alpha, int relu)
{
  const int wave = threadIdx.x >> 5;
  const int lane = threadIdx.x & 31;
  const int t = blockIdx.x * 4 + wave;
  if (t >= Mtiles * Ntiles) return;
  const int mt = t / Ntiles;
  const int nt = t - mt * Ntiles;
  const long long z = blockIdx.z;

  const int half = lane >> 4;   // K-half select (ISA bf16 A/B fragment layout)
  const int idx  = lane & 15;   // A row / B column within tile

  const __bf16* ap = A  + z * sA + (long long)(mt * 16 + idx) * lda + half * 8;
  const __bf16* bp = BT + z * sB + (long long)(nt * 16 + idx) * ldb + half * 8;

  v8f acc = {};
  for (int kt = 0; kt < kTiles; ++kt) {
    BfFrag fa, fb;
    fa.q[0] = *(const u32x4*)(ap);        // K = kbase + half*8 .. +7
    fa.q[1] = *(const u32x4*)(ap + 16);   // K = kbase + 16 + half*8 .. +7
    fb.q[0] = *(const u32x4*)(bp);
    fb.q[1] = *(const u32x4*)(bp + 16);
    ap += 32; bp += 32;
    if (kt + 1 < kTiles) {                // gfx1250 global_prefetch_b8
      __builtin_prefetch((const void*)ap, 0, 3);
      __builtin_prefetch((const void*)bp, 0, 3);
    }
    acc = __builtin_amdgcn_wmma_f32_16x16x32_bf16(
        false, fa.v, false, fb.v, (short)0, acc, false, false);
  }

  const int gn = nt * 16 + idx;
  const bool ncol = (gn < Nreal);
  const float bv = (bias && ncol) ? bias[gn] : 0.0f;
  const int rbase = mt * 16 + half * 8;   // C/D layout: lane half selects M 0-7 / 8-15
#pragma unroll
  for (int i = 0; i < 8; ++i) {
    const int gm = rbase + i;
    if (gm < Mreal && ncol) {
      float f = acc[i] * alpha + bv;
      if (relu) f = fmaxf(f, 0.0f);
      if (outf) outf[z * sO  + (long long)gm * ldo  + gn] = f;
      if (outb) outb[z * sOb + (long long)gm * ldob + gn] = (__bf16)f;
    }
  }
}

// =====================================================================
// V2: LDS-staged WMMA GEMM with CDNA5 async loads, double-buffered.
// Block = 256 threads (8 waves) computes a 128x64 output tile.
// Each wave: 16 rows x 64 cols = 4 accumulators; A fragment reused 4x.
// Requires: M % 128 == 0, N % 64 == 0, K = kTiles*32.
// =====================================================================
__global__ __launch_bounds__(256) void gemm_wmma_lds(
    const __bf16* __restrict__ A,  int lda,
    const __bf16* __restrict__ BT, int ldb,
    float*  __restrict__ outf, int ldo,
    __bf16* __restrict__ outb, int ldob,
    const float* __restrict__ bias,
    int kTiles, float alpha, int relu)
{
  __shared__ __bf16 sA[2][128][32];   // 16 KB
  __shared__ __bf16 sB[2][64][32];    // 8 KB

  const int m0 = blockIdx.x * 128;
  const int n0 = blockIdx.y * 64;
  const int tid = threadIdx.x;
  const int lane = tid & 31, wv = tid >> 5;
  const int half = lane >> 4, idx = lane & 15;

  // Issue one K-chunk (32 wide) of A(128 rows) + B(64 rows) into buffer p.
  // A: 512 x 16B segments (2 per thread), B: 256 x 16B segments (1 per thread).
  auto issue = [&](int kt, int p) {
    const int kb = kt * 32;
    {
      int s = tid;                  // A segment 0
      int row = s >> 2, sg = (s & 3) * 8;
      async_load_b128((unsigned)(uintptr_t)&sA[p][row][sg],
                      A + (long long)(m0 + row) * lda + kb + sg);
    }
    {
      int s = tid + 256;            // A segment 1
      int row = s >> 2, sg = (s & 3) * 8;
      async_load_b128((unsigned)(uintptr_t)&sA[p][row][sg],
                      A + (long long)(m0 + row) * lda + kb + sg);
    }
    {
      int row = tid >> 2, sg = (tid & 3) * 8;   // B segment
      async_load_b128((unsigned)(uintptr_t)&sB[p][row][sg],
                      BT + (long long)(n0 + row) * ldb + kb + sg);
    }
  };

  v8f acc[4] = {{}, {}, {}, {}};
  issue(0, 0);

  for (int kt = 0; kt < kTiles; ++kt) {
    const int cur = kt & 1;
    if (kt + 1 < kTiles) { issue(kt + 1, cur ^ 1); wait_async_le3(); }
    else                 { wait_async_le0(); }
    __syncthreads();

    const __bf16* ar = &sA[cur][wv * 16 + idx][half * 8];
    BfFrag fa;
    fa.q[0] = *(const u32x4*)(ar);
    fa.q[1] = *(const u32x4*)(ar + 16);
#pragma unroll
    for (int ns = 0; ns < 4; ++ns) {
      const __bf16* br = &sB[cur][ns * 16 + idx][half * 8];
      BfFrag fb;
      fb.q[0] = *(const u32x4*)(br);
      fb.q[1] = *(const u32x4*)(br + 16);
      acc[ns] = __builtin_amdgcn_wmma_f32_16x16x32_bf16(
          false, fa.v, false, fb.v, (short)0, acc[ns], false, false);
    }
    __syncthreads();   // protect buffer `cur` before it is re-filled
  }

  const int gmb = m0 + wv * 16 + half * 8;
#pragma unroll
  for (int ns = 0; ns < 4; ++ns) {
    const int gn = n0 + ns * 16 + idx;
    const float bv = bias ? bias[gn] : 0.0f;
#pragma unroll
    for (int i = 0; i < 8; ++i) {
      float f = acc[ns][i] * alpha + bv;
      if (relu) f = fmaxf(f, 0.0f);
      const long long o = (long long)(gmb + i);
      if (outf) outf[o * ldo  + gn] = f;
      if (outb) outb[o * ldob + gn] = (__bf16)f;
    }
  }
}

// =====================================================================
// Packing / repacking kernels
// =====================================================================
__global__ void k_f32_to_bf16(const float* __restrict__ s, __bf16* __restrict__ d, int n) {
  int i = blockIdx.x * blockDim.x + threadIdx.x;
  if (i < n) d[i] = (__bf16)s[i];
}

// mv = concat(ma[B,60,E], veh[B,40,E]) -> bf16 [B*100, 256]
__global__ void k_pack_mv(const float* __restrict__ ma, const float* __restrict__ veh,
                          __bf16* __restrict__ d) {
  int i = blockIdx.x * blockDim.x + threadIdx.x;
  if (i >= BC * NE) return;
  int e = i & (NE - 1); int row = i >> 8;
  int c = row % NC;     int b = row / NC;
  float v = (c < NMM) ? ma[((long long)(b * NMM + c)) * NE + e]
                      : veh[((long long)(b * NVV + (c - NMM))) * NE + e];
  d[i] = (__bf16)v;
}

// W[K,N] f32 -> WT[N,K] bf16
__global__ void k_transpose_bf16(const float* __restrict__ W, __bf16* __restrict__ WT,
                                 int K, int N) {
  int i = blockIdx.x * blockDim.x + threadIdx.x;
  if (i >= K * N) return;
  int k = i % K; int n = i / K;
  WT[i] = (__bf16)W[(long long)k * N + n];
}

// q_rm bf16 [B*R,256] -> head-packed [BH][RP][32], K padded 16->32 with zeros
__global__ void k_repack_q(const __bf16* __restrict__ q, __bf16* __restrict__ qh) {
  int i = blockIdx.x * blockDim.x + threadIdx.x;
  if (i >= BH * RP * 32) return;
  int dd = i & 31; int r = (i >> 5) % RP; int bh = i / (RP * 32);
  int b = bh >> 4, h = bh & 15;
  __bf16 v = (__bf16)0.0f;
  if (r < NR && dd < ND) v = q[((long long)(b * NR + r)) * NHD + h * ND + dd];
  qh[i] = v;
}

// k_rm bf16 [B*C,256] -> head-packed [BH][CP][32]
__global__ void k_repack_k(const __bf16* __restrict__ k, __bf16* __restrict__ kh) {
  int i = blockIdx.x * blockDim.x + threadIdx.x;
  if (i >= BH * CP * 32) return;
  int dd = i & 31; int c = (i >> 5) % CP; int bh = i / (CP * 32);
  int b = bh >> 4, h = bh & 15;
  __bf16 v = (__bf16)0.0f;
  if (c < NC && dd < ND) v = k[((long long)(b * NC + c)) * NHD + h * ND + dd];
  kh[i] = v;
}

// v_rm bf16 [B*C,256] -> vT [BH][16][CW]  (vT[d][c] = v[c, h*16+d]), c padded ->128 zeros
__global__ void k_repack_vT(const __bf16* __restrict__ v, __bf16* __restrict__ vt) {
  int i = blockIdx.x * blockDim.x + threadIdx.x;
  if (i >= BH * ND * CW) return;
  int c = i & (CW - 1); int dd = (i >> 7) & 15; int bh = i >> 11;
  int b = bh >> 4, h = bh & 15;
  __bf16 val = (__bf16)0.0f;
  if (c < NC) val = v[((long long)(b * NC + c)) * NHD + h * ND + dd];
  vt[i] = val;
}

// attn f32 [BH][RP][16] -> out_concat bf16 [B*R,256]
__global__ void k_merge_heads(const float* __restrict__ attn, __bf16* __restrict__ oc) {
  int i = blockIdx.x * blockDim.x + threadIdx.x;
  if (i >= BR * NE) return;
  int col = i & (NE - 1); int row = i >> 8;
  int h = col >> 4; int dd = col & 15;
  int b = row / NR; int r = row % NR;
  oc[i] = (__bf16)attn[((long long)(b * NH + h) * RP + r) * ND + dd];
}

// =====================================================================
// Mix-MLP (per-head 2->16->1) + softmax over C.  dot is read then
// overwritten in-place with mixed scores; writes bf16 w [BH][RP][CW].
// =====================================================================
__global__ __launch_bounds__(256) void k_mix_softmax(
    const float* __restrict__ proc, const float* __restrict__ trans,
    float* __restrict__ dot, __bf16* __restrict__ wbf,
    const float* __restrict__ m1w, const float* __restrict__ m1b,
    const float* __restrict__ m2w, const float* __restrict__ m2b)
{
  const int bh = blockIdx.x;
  const int b = bh >> 4, h = bh & 15;
  const int r = threadIdx.x;
  if (r >= RP) return;
  __bf16* wrow = wbf + ((long long)bh * RP + r) * CW;
  if (r >= NR) {                       // zero A-pad rows for the w@V GEMM
    for (int c = 0; c < CW; ++c) wrow[c] = (__bf16)0.0f;
    return;
  }
  float w10[16], w11[16], b1[16], w2[16];
#pragma unroll
  for (int m = 0; m < 16; ++m) {
    w10[m] = m1w[h * 32 + m];
    w11[m] = m1w[h * 32 + 16 + m];
    b1[m]  = m1b[h * 16 + m];
    w2[m]  = m2w[h * 16 + m];
  }
  const float b2 = m2b[h];
  float* drow = dot + ((long long)bh * NR + r) * CP;
  const float* prow = proc  + ((long long)(b * NR) + r) * NMM;
  const float* trow = trans + ((long long)(b * NR) + r) * NVV;

  float mx = -1e30f;
  for (int c = 0; c < NC; ++c) {
    float dv = drow[c];
    float cv = (c < NMM) ? prow[c] : trow[c - NMM];
    float acc = b2;
#pragma unroll
    for (int m = 0; m < 16; ++m) {
      float s = dv * w10[m] + cv * w11[m] + b1[m];
      acc += fmaxf(s, 0.0f) * w2[m];
    }
    drow[c] = acc;
    mx = fmaxf(mx, acc);
  }
  float sum = 0.0f;
  for (int c = 0; c < NC; ++c) sum += __expf(drow[c] - mx);
  const float inv = 1.0f / sum;
  for (int c = 0; c < NC; ++c) wrow[c] = (__bf16)(__expf(drow[c] - mx) * inv);
  for (int c = NC; c < CW; ++c) wrow[c] = (__bf16)0.0f;
}

// =====================================================================
// Fused residual + InstanceNorm1d over sequence dim (per b, per channel e).
// =====================================================================
__global__ __launch_bounds__(256) void k_inorm(
    const float* __restrict__ x1, const float* __restrict__ x2,
    const float* __restrict__ g, const float* __restrict__ bb,
    float* __restrict__ outf, __bf16* __restrict__ outb)
{
  const int b = blockIdx.x;
  const int e = threadIdx.x;
  const float* p1 = x1 + (long long)b * NR * NE + e;
  const float* p2 = x2 + (long long)b * NR * NE + e;
  float sum = 0.0f, sq = 0.0f;
  for (int r = 0; r < NR; ++r) {
    float v = p1[r * NE] + p2[r * NE];
    sum += v; sq += v * v;
  }
  const float mean = sum * (1.0f / NR);
  const float var  = sq  * (1.0f / NR) - mean * mean;
  const float rs   = rsqrtf(var + 1e-5f);
  const float ge = g[e], be = bb[e];
  for (int r = 0; r < NR; ++r) {
    float v = (p1[r * NE] + p2[r * NE] - mean) * rs * ge + be;
    long long o = (long long)b * NR * NE + (long long)r * NE + e;
    if (outf) outf[o] = v;
    if (outb) outb[o] = (__bf16)v;
  }
}

// =====================================================================
// Host launcher
// =====================================================================
static inline int ceil_div(int a, int b) { return (a + b - 1) / b; }

extern "C" void kernel_launch(void* const* d_in, const int* in_sizes, int n_in,
                              void* d_out, int out_size, void* d_ws, size_t ws_size,
                              hipStream_t stream)
{
  (void)in_sizes; (void)n_in; (void)out_size; (void)ws_size;
  const float* ope   = (const float*)d_in[0];
  const float* ma    = (const float*)d_in[1];
  const float* veh   = (const float*)d_in[2];
  const float* proc  = (const float*)d_in[3];
  const float* trans = (const float*)d_in[4];
  const float* Wq    = (const float*)d_in[5];
  const float* Wk    = (const float*)d_in[6];
  const float* Wv    = (const float*)d_in[7];
  const float* m1w   = (const float*)d_in[8];
  const float* m1b   = (const float*)d_in[9];
  const float* m2w   = (const float*)d_in[10];
  const float* m2b   = (const float*)d_in[11];
  const float* mhcw  = (const float*)d_in[12];
  const float* mhcb  = (const float*)d_in[13];
  const float* in1g  = (const float*)d_in[14];
  const float* in1b  = (const float*)d_in[15];
  const float* ffw1  = (const float*)d_in[16];
  const float* ffb1  = (const float*)d_in[17];
  const float* ffw2  = (const float*)d_in[18];
  const float* ffb2  = (const float*)d_in[19];
  const float* in2g  = (const float*)d_in[20];
  const float* in2b  = (const float*)d_in[21];
  float* outp = (float*)d_out;

  // ---- workspace carve-out (256B aligned) ----
  size_t off = 0;
  auto alloc = [&](size_t bytes) -> void* {
    off = (off + 255) & ~(size_t)255;
    void* p = (char*)d_ws + off;
    off += bytes;
    return p;
  };
  __bf16* ope_bf  = (__bf16*)alloc((size_t)BR * NE * 2);
  __bf16* mv_bf   = (__bf16*)alloc((size_t)BC * NE * 2);
  __bf16* WqT     = (__bf16*)alloc((size_t)NE * NHD * 2);
  __bf16* WkT     = (__bf16*)alloc((size_t)NE * NHD * 2);
  __bf16* WvT     = (__bf16*)alloc((size_t)NE * NHD * 2);
  __bf16* mhcT    = (__bf16*)alloc((size_t)NHD * NE * 2);
  __bf16* ffw1T   = (__bf16*)alloc((size_t)NF * NE * 2);
  __bf16* ffw2T   = (__bf16*)alloc((size_t)NE * NF * 2);
  __bf16* q_rm    = (__bf16*)alloc((size_t)BR * NHD * 2);
  __bf16* k_rm    = (__bf16*)alloc((size_t)BC * NHD * 2);
  __bf16* v_rm    = (__bf16*)alloc((size_t)BC * NHD * 2);
  __bf16* qhp     = (__bf16*)alloc((size_t)BH * RP * 32 * 2);
  __bf16* khp     = (__bf16*)alloc((size_t)BH * CP * 32 * 2);
  __bf16* vT      = (__bf16*)alloc((size_t)BH * ND * CW * 2);
  float*  dotbuf  = (float*) alloc((size_t)BH * NR * CP * 4);
  __bf16* wbf     = (__bf16*)alloc((size_t)BH * RP * CW * 2);
  float*  attn    = (float*) alloc((size_t)BH * RP * ND * 4);
  __bf16* ocbf    = (__bf16*)alloc((size_t)BR * NE * 2);
  float*  mhf     = (float*) alloc((size_t)BR * NE * 4);
  float*  out1f   = (float*) alloc((size_t)BR * NE * 4);
  __bf16* out1bf  = (__bf16*)alloc((size_t)BR * NE * 2);
  __bf16* ffhbf   = (__bf16*)alloc((size_t)BR * NF * 2);
  float*  ff2f    = (float*) alloc((size_t)BR * NE * 4);

  const dim3 blk256(256);
  const dim3 gblk(128);

  // ---- pack inputs ----
  k_f32_to_bf16<<<ceil_div(BR * NE, 256), blk256, 0, stream>>>(ope, ope_bf, BR * NE);
  k_pack_mv<<<ceil_div(BC * NE, 256), blk256, 0, stream>>>(ma, veh, mv_bf);
  k_transpose_bf16<<<ceil_div(NE * NHD, 256), blk256, 0, stream>>>(Wq,  WqT,  NE, NHD);
  k_transpose_bf16<<<ceil_div(NE * NHD, 256), blk256, 0, stream>>>(Wk,  WkT,  NE, NHD);
  k_transpose_bf16<<<ceil_div(NE * NHD, 256), blk256, 0, stream>>>(Wv,  WvT,  NE, NHD);
  k_transpose_bf16<<<ceil_div(NHD * NE, 256), blk256, 0, stream>>>(mhcw, mhcT, NHD, NE);
  k_transpose_bf16<<<ceil_div(NE * NF, 256), blk256, 0, stream>>>(ffw1, ffw1T, NE, NF);
  k_transpose_bf16<<<ceil_div(NF * NE, 256), blk256, 0, stream>>>(ffw2, ffw2T, NF, NE);

  // ---- Q/K/V projections (bf16 out), LDS/async GEMM ----
  gemm_wmma_lds<<<dim3(BR / 128, NHD / 64), blk256, 0, stream>>>(
      ope_bf, NE, WqT, NE, nullptr, 0, q_rm, NHD, nullptr, NE / 32, 1.0f, 0);
  gemm_wmma_lds<<<dim3(BC / 128, NHD / 64), blk256, 0, stream>>>(
      mv_bf, NE, WkT, NE, nullptr, 0, k_rm, NHD, nullptr, NE / 32, 1.0f, 0);
  gemm_wmma_lds<<<dim3(BC / 128, NHD / 64), blk256, 0, stream>>>(
      mv_bf, NE, WvT, NE, nullptr, 0, v_rm, NHD, nullptr, NE / 32, 1.0f, 0);

  // ---- head repack (zero-padded K and rows) ----
  k_repack_q<<<ceil_div(BH * RP * 32, 256), blk256, 0, stream>>>(q_rm, qhp);
  k_repack_k<<<ceil_div(BH * CP * 32, 256), blk256, 0, stream>>>(k_rm, khp);
  k_repack_vT<<<ceil_div(BH * ND * CW, 256), blk256, 0, stream>>>(v_rm, vT);

  // ---- dot = (q @ k^T) / sqrt(D), batched over B*H (direct WMMA) ----
  {
    int mt = RP / 16, nt = CP / 16;                // 13 x 7
    gemm_wmma_bf16<<<dim3(ceil_div(mt * nt, 4), 1, BH), gblk, 0, stream>>>(
        qhp, 32, (long long)RP * 32, khp, 32, (long long)CP * 32,
        dotbuf, CP, (long long)NR * CP, nullptr, 0, 0,
        nullptr, mt, nt, 1, NR, NC, 0.25f, 0);
  }

  // ---- mix MLP + softmax -> bf16 attention weights ----
  k_mix_softmax<<<dim3(BH), blk256, 0, stream>>>(proc, trans, dotbuf, wbf,
                                                 m1w, m1b, m2w, m2b);

  // ---- out = w @ v, batched (direct WMMA) ----
  {
    int mt = RP / 16, nt = 1;                      // 13 x 1
    gemm_wmma_bf16<<<dim3(ceil_div(mt * nt, 4), 1, BH), gblk, 0, stream>>>(
        wbf, CW, (long long)RP * CW, vT, CW, (long long)ND * CW,
        attn, ND, (long long)RP * ND, nullptr, 0, 0,
        nullptr, mt, nt, CW / 32, NR, ND, 1.0f, 0);
  }
  k_merge_heads<<<ceil_div(BR * NE, 256), blk256, 0, stream>>>(attn, ocbf);

  // ---- mh = out_concat @ mhc_w + mhc_b ----
  gemm_wmma_lds<<<dim3(BR / 128, NE / 64), blk256, 0, stream>>>(
      ocbf, NHD, mhcT, NHD, mhf, NE, nullptr, 0, mhcb, NHD / 32, 1.0f, 0);

  // ---- out1 = InstanceNorm(ope + mh) ----
  k_inorm<<<dim3(NB), blk256, 0, stream>>>(ope, mhf, in1g, in1b, out1f, out1bf);

  // ---- ffh = relu(out1 @ ff_w1 + b1) (bf16 out) ----
  gemm_wmma_lds<<<dim3(BR / 128, NF / 64), blk256, 0, stream>>>(
      out1bf, NE, ffw1T, NE, nullptr, 0, ffhbf, NF, ffb1, NE / 32, 1.0f, 1);

  // ---- ff2 = ffh @ ff_w2 + b2 ----
  gemm_wmma_lds<<<dim3(BR / 128, NE / 64), blk256, 0, stream>>>(
      ffhbf, NF, ffw2T, NF, ff2f, NE, nullptr, 0, ffb2, NF / 32, 1.0f, 0);

  // ---- final: InstanceNorm(out1 + ff2) -> d_out ----
  k_inorm<<<dim3(NB), blk256, 0, stream>>>(out1f, ff2f, in2g, in2b, outp, nullptr);
}
